// NeighborhoodSelfAttention_12781822672999
// MI455X (gfx1250) — matrix operand order repaired and checked by reading, not compiled, not run on hardware
//
#include <hip/hip_runtime.h>

// ---- CDNA5 / gfx1250: wave32, WMMA 16x16x32 f16 -> f32, async LDS loads ----

typedef __attribute__((ext_vector_type(16))) _Float16 v16h;
typedef __attribute__((ext_vector_type(8)))  float    v8f;
typedef __attribute__((ext_vector_type(4)))  int      i4;
typedef __attribute__((ext_vector_type(2)))  int      i2;
typedef __attribute__((ext_vector_type(4)))  float    f4;

#define N_NODES 50000
#define DDIM    128
#define KNBR    16
#define HEADS   8

union V16 { v16h h; i4 q[2]; };

static __device__ inline unsigned pack_f16x2(float a, float b) {
  union { _Float16 f; unsigned short s; } x, y;
  x.f = (_Float16)a; y.f = (_Float16)b;
  return (unsigned)x.s | ((unsigned)y.s << 16);
}

// ------------------------------------------------------------------
// prep: weights (f32 row-major 128x128) -> f16 WMMA-B panels.
// Panel(mat,c,T) is the 32(K) x 16(col) block, dims 32c..32c+31,
// cols 16T..16T+15, in B register layout:
//   lane (j,hi) : VGPR r (packed 2 f16) = W[32c+16(r>>2)+8hi+2(r&3) (+1)][16T+j]
// Stored lane-major, 8 uints per lane:
//   Wp[(((mat*4+c)*8+T)*32+lane)*8 + r]
// ------------------------------------------------------------------
__global__ __launch_bounds__(256) void prep_panels(
    const float* __restrict__ Wq, const float* __restrict__ Wk,
    const float* __restrict__ Wv, const float* __restrict__ Wo,
    unsigned* __restrict__ Wp) {
  int tid  = blockIdx.x * blockDim.x + threadIdx.x;  // 0..4095
  int mat  = tid >> 10;
  int c    = (tid >> 8) & 3;
  int T    = (tid >> 5) & 7;
  int lane = tid & 31;
  int j = lane & 15, hi = lane >> 4;
  const float* W = (mat == 0) ? Wq : (mat == 1) ? Wk : (mat == 2) ? Wv : Wo;
  int col = T * 16 + j;
  unsigned* dst = Wp + (((mat * 4 + c) * 8 + T) * 32 + lane) * 8;
#pragma unroll
  for (int r = 0; r < 8; ++r) {
    int k0 = c * 32 + (r >> 2) * 16 + hi * 8 + 2 * (r & 3);
    dst[r] = pack_f16x2(W[k0 * DDIM + col], W[(k0 + 1) * DDIM + col]);
  }
}

// ------------------------------------------------------------------
// Fused Q/K/V projection: C = x @ W + b for 3 weights, f16 outputs.
// Block = 4 waves, 16-row tile; wave w owns cols [32w,32w+32).
// A (16x32 f16) per ISA layout: lane (m,hi) holds dims
//   {32c+8hi+0..7, 32c+16+8hi+0..7} of row m.
// ------------------------------------------------------------------
__global__ __launch_bounds__(128) void qkv_proj(
    const float* __restrict__ x, const unsigned* __restrict__ Wp,
    const float* __restrict__ bq, const float* __restrict__ bk,
    const float* __restrict__ bv,
    _Float16* __restrict__ qo, _Float16* __restrict__ ko,
    _Float16* __restrict__ vo) {
  const int wave = threadIdx.x >> 5;
  const int lane = threadIdx.x & 31;
  const int m = lane & 15, hi = lane >> 4;
  const int row0 = blockIdx.x * 16;

  v8f acc[3][2] = {};
#pragma unroll
  for (int c = 0; c < 4; ++c) {
    const float* xr = x + (size_t)(row0 + m) * DDIM + c * 32 + hi * 8;
    f4 f0 = *(const f4*)(xr);
    f4 f1 = *(const f4*)(xr + 4);
    f4 f2 = *(const f4*)(xr + 16);
    f4 f3 = *(const f4*)(xr + 20);
    V16 a;
#pragma unroll
    for (int i = 0; i < 4; ++i) {
      a.h[i]      = (_Float16)f0[i];
      a.h[4 + i]  = (_Float16)f1[i];
      a.h[8 + i]  = (_Float16)f2[i];
      a.h[12 + i] = (_Float16)f3[i];
    }
#pragma unroll
    for (int t = 0; t < 2; ++t) {
      int T = wave * 2 + t;
#pragma unroll
      for (int mat = 0; mat < 3; ++mat) {
        V16 b;
        const i4* bp = (const i4*)(Wp + (((mat * 4 + c) * 8 + T) * 32 + lane) * 8);
        b.q[0] = bp[0];
        b.q[1] = bp[1];
        acc[mat][t] = __builtin_amdgcn_wmma_f32_16x16x32_f16(
            false, a.h, false, b.h, (short)0, acc[mat][t], false, false);
      }
    }
  }
  // D layout: lane holds col = m, rows r+8*hi. Add bias, store f16.
#pragma unroll
  for (int t = 0; t < 2; ++t) {
    int col = wave * 32 + t * 16 + m;
    float bqs = bq[col], bks = bk[col], bvs = bv[col];
#pragma unroll
    for (int r = 0; r < 8; ++r) {
      size_t o = (size_t)(row0 + r + 8 * hi) * DDIM + col;
      qo[o] = (_Float16)(acc[0][t][r] + bqs);
      ko[o] = (_Float16)(acc[1][t][r] + bks);
      vo[o] = (_Float16)(acc[2][t][r] + bvs);
    }
  }
}

// ------------------------------------------------------------------
// Attention core: one wave per node.
// Scores via WMMA with block-diagonal A: A[h,d]=q[d] iff d/16==h, so
// D[h,k] = dot(q_h, Kg[k]_h). B = gathered K rows in B layout
// (lane (j,hi): dims 32c+8hi+0..7, 32c+16+8hi+0..7 of row idx[j]).
// Gathered V staged to LDS via GLOBAL_LOAD_ASYNC_TO_LDS_B128
// (ASYNCcnt path); softmax across 16 column-lanes via shuffles.
// ------------------------------------------------------------------
__global__ __launch_bounds__(256) void attn_kernel(
    const int* __restrict__ neighbors, const _Float16* __restrict__ qf,
    const _Float16* __restrict__ kf, const _Float16* __restrict__ vf,
    _Float16* __restrict__ att) {
  __shared__ unsigned short vsh[8 * 16 * DDIM];  // 8 waves * 16 rows * 128 f16
  __shared__ float ash[8 * HEADS * 16];          // attn weights per wave

  const int wave = threadIdx.x >> 5;
  const int lane = threadIdx.x & 31;
  const int m = lane & 15, hi = lane >> 4;
  const int n = blockIdx.x * 8 + wave;

  int nb = neighbors[n * KNBR + m];
  bool valid = nb >= 0;
  int row = valid ? nb : 0;  // matches reference's clip(neighbors, 0)

  // Kick off async memory->LDS staging of gathered V rows first so the
  // DMA overlaps the K-gather + WMMA + softmax below.
  // 16 rows x 16 b128-chunks, 8 per lane. GV mode: VDST = LDS byte
  // offset (low 32 bits of the generic shared pointer), VADDR = 64-bit
  // global address.
  unsigned short* vw = vsh + wave * 16 * DDIM;
#pragma unroll
  for (int it = 0; it < 8; ++it) {
    int cid = it * 32 + lane;
    int r = cid >> 4, cc = cid & 15;
    int rrow = __shfl(row, r, 32);
    unsigned lds_a = (unsigned)(uintptr_t)(vw + cid * 8);
    const void* ga = (const void*)(vf + (size_t)rrow * DDIM + cc * 8);
    asm volatile("global_load_async_to_lds_b128 %0, %1, off"
                 :: "v"(lds_a), "v"(ga) : "memory");
  }

  // q fragment: bytes [32(m&7)+16hi, +16) of q row n covers exactly the
  // nonzero block-diagonal entries this lane ever contributes.
  i4 qv = *(const i4*)(qf + (size_t)n * DDIM + (m & 7) * 16 + hi * 8);
  i4 z4 = (i4)0;

  // Gathered K rows as 4 WMMA B operands (2 x b128 per lane per chunk).
  V16 bop[4];
#pragma unroll
  for (int c = 0; c < 4; ++c) {
    const _Float16* kp = kf + (size_t)row * DDIM + c * 32 + hi * 8;
    bop[c].q[0] = *(const i4*)(kp);
    bop[c].q[1] = *(const i4*)(kp + 16);
  }

  v8f sc = {};
#pragma unroll
  for (int c = 0; c < 4; ++c) {
    V16 a;
    a.q[0] = (m == 2 * c)     ? qv : z4;  // rows = heads 2c
    a.q[1] = (m == 2 * c + 1) ? qv : z4;  // rows = heads 2c+1
    sc = __builtin_amdgcn_wmma_f32_16x16x32_f16(
        false, a.h, false, bop[c].h, (short)0, sc, false, false);
  }

  // D: VGPR r, lanes 0-15 -> score[head r][neighbor m]. Mask + softmax
  // across the 16 column lanes (xor masks 1,2,4,8 stay in each half).
  float attnv[8];
#pragma unroll
  for (int r = 0; r < 8; ++r) {
    float s = valid ? sc[r] * 0.25f : -1.0e9f;  // 1/sqrt(16)
    float mx = s;
#pragma unroll
    for (int d = 1; d < 16; d <<= 1) mx = fmaxf(mx, __shfl_xor(mx, d, 32));
    float e = __expf(s - mx);
    float sum = e;
#pragma unroll
    for (int d = 1; d < 16; d <<= 1) sum += __shfl_xor(sum, d, 32);
    attnv[r] = e / sum;
  }
  float* aw = ash + wave * HEADS * 16;
  if (lane < 16) {
#pragma unroll
    for (int r = 0; r < 8; ++r) aw[r * 16 + m] = attnv[r];
  }

  // Drain this wave's async V staging, then block barrier (also covers
  // the attn ds_stores above).
  asm volatile("s_wait_asynccnt 0x0" ::: "memory");
  __syncthreads();

  // attended[h,d] = sum_k attn[h,k] * Vg[k,d]; lane owns dims 4L..4L+3,
  // which lie in head L>>2.
  const int d0 = lane * 4;
  const int h = lane >> 2;
  float a0 = 0.f, a1 = 0.f, a2 = 0.f, a3 = 0.f;
#pragma unroll
  for (int k = 0; k < 16; ++k) {
    float a = aw[h * 16 + k];
    const _Float16* vp = (const _Float16*)(vw + k * DDIM + d0);
    a0 += a * (float)vp[0];
    a1 += a * (float)vp[1];
    a2 += a * (float)vp[2];
    a3 += a * (float)vp[3];
  }
  i2 o;
  o[0] = (int)pack_f16x2(a0, a1);
  o[1] = (int)pack_f16x2(a2, a3);
  *(i2*)(att + (size_t)n * DDIM + d0) = o;
}

// ------------------------------------------------------------------
// Output projection: out = attended @ Wo + bo (f32 out).
// ------------------------------------------------------------------
__global__ __launch_bounds__(128) void out_proj(
    const _Float16* __restrict__ att, const unsigned* __restrict__ Wp,
    const float* __restrict__ bo, float* __restrict__ out) {
  const int wave = threadIdx.x >> 5;
  const int lane = threadIdx.x & 31;
  const int m = lane & 15, hi = lane >> 4;
  const int row0 = blockIdx.x * 16;

  v8f acc[2] = {};
#pragma unroll
  for (int c = 0; c < 4; ++c) {
    const _Float16* ar = att + (size_t)(row0 + m) * DDIM + c * 32 + hi * 8;
    V16 a;
    a.q[0] = *(const i4*)(ar);
    a.q[1] = *(const i4*)(ar + 16);
#pragma unroll
    for (int t = 0; t < 2; ++t) {
      int T = wave * 2 + t;
      V16 b;
      const i4* bp = (const i4*)(Wp + (((3 * 4 + c) * 8 + T) * 32 + lane) * 8);
      b.q[0] = bp[0];
      b.q[1] = bp[1];
      acc[t] = __builtin_amdgcn_wmma_f32_16x16x32_f16(
          false, a.h, false, b.h, (short)0, acc[t], false, false);
    }
  }
#pragma unroll
  for (int t = 0; t < 2; ++t) {
    int col = wave * 32 + t * 16 + m;
    float b = bo[col];
#pragma unroll
    for (int r = 0; r < 8; ++r)
      out[(size_t)(row0 + r + 8 * hi) * DDIM + col] = acc[t][r] + b;
  }
}

extern "C" void kernel_launch(void* const* d_in, const int* in_sizes, int n_in,
                              void* d_out, int out_size, void* d_ws,
                              size_t ws_size, hipStream_t stream) {
  const float* x         = (const float*)d_in[0];
  const int*   neighbors = (const int*)d_in[1];
  const float* Wq = (const float*)d_in[2];
  const float* bq = (const float*)d_in[3];
  const float* Wk = (const float*)d_in[4];
  const float* bk = (const float*)d_in[5];
  const float* Wv = (const float*)d_in[6];
  const float* bv = (const float*)d_in[7];
  const float* Wo = (const float*)d_in[8];
  const float* bo = (const float*)d_in[9];
  float* out = (float*)d_out;

  char* ws = (char*)d_ws;
  unsigned* Wp = (unsigned*)ws;  // 4 mats * 32 panels * 1KB = 128KB
  size_t off = 131072;
  const size_t mat_bytes = (size_t)N_NODES * DDIM * sizeof(_Float16);
  _Float16* qf  = (_Float16*)(ws + off); off += mat_bytes;
  _Float16* kf  = (_Float16*)(ws + off); off += mat_bytes;
  _Float16* vf  = (_Float16*)(ws + off); off += mat_bytes;
  _Float16* atf = (_Float16*)(ws + off); off += mat_bytes;
  (void)ws_size; (void)in_sizes; (void)n_in; (void)out_size;

  prep_panels<<<16, 256, 0, stream>>>(Wq, Wk, Wv, Wo, Wp);
  qkv_proj<<<N_NODES / 16, 128, 0, stream>>>(x, Wp, bq, bk, bv, qf, kf, vf);
  attn_kernel<<<N_NODES / 8, 256, 0, stream>>>(neighbors, qf, kf, vf, atf);
  out_proj<<<N_NODES / 16, 128, 0, stream>>>(atf, Wp, bo, out);
}
